// GraphAttentionMoudle_61615600828785
// MI455X (gfx1250) — compile-verified, matrix operand order checked by reading, not run on hardware
//
#include <hip/hip_runtime.h>
#include <hip/hip_bf16.h>

#define HH   4
#define NN   4096
#define DHH  128
#define HD   512          // H * DH
#define GAT_ALPHA 0.2f
#define NEGV (-9.0e15f)

typedef _Float16 v16h  __attribute__((ext_vector_type(16)));
typedef _Float16 half8 __attribute__((ext_vector_type(8)));
typedef float    v8f   __attribute__((ext_vector_type(8)));

// 16-bit A/B fragment K mapping (ISA 7.12.2): lanes 0-15 kbase=0, lanes 16-31 kbase=8
// half g<8  -> K = kbase + g ; half g>=8 -> K = kbase + 16 + (g-8) = kbase + 8 + g
__device__ __forceinline__ int kmap(int kbase, int g) {
  return (g < 8) ? (kbase + g) : (kbase + 8 + g);
}

__device__ __forceinline__ v16h make_v16h(half8 lo, half8 hi) {
  return __builtin_shufflevector(lo, hi, 0,1,2,3,4,5,6,7,8,9,10,11,12,13,14,15);
}

// ---------------------------------------------------------------------------
// Pack adjacency int32 [N][N] -> bitmask [N][N/32] (bit j of word jb = adj!=0).
// Reads the 64MB adj exactly once; all later passes read 2MB of bits from L2.
// ---------------------------------------------------------------------------
__global__ void pack_adj_kernel(const int* __restrict__ adj, unsigned* __restrict__ bits) {
  int tid = blockIdx.x * blockDim.x + threadIdx.x;   // row*(N/32) + jb
  if (tid >= NN * (NN / 32)) return;
  const int* p = adj + (size_t)tid * 32;
  unsigned w = 0u;
#pragma unroll
  for (int g = 0; g < 32; ++g) w |= (p[g] != 0 ? 1u : 0u) << g;
  bits[tid] = w;
}

// ---------------------------------------------------------------------------
// Convert W (f32, [H][Kdim][128]) to WMMA-B fragment-swizzled f16:
// slot = ((h*nks + ks)*8 + nt), each lane's 16 halves stored contiguously.
// ---------------------------------------------------------------------------
__global__ void prep_w_kernel(const float* __restrict__ W, _Float16* __restrict__ dst,
                              int Kdim) {
  int tid  = blockIdx.x * blockDim.x + threadIdx.x;
  int lane = tid & 31;
  int slot = tid >> 5;
  int nks  = Kdim >> 5;
  int nt   = slot & 7;
  int hk   = slot >> 3;
  int ks   = hk % nks;
  int h    = hk / nks;
  if (h >= HH) return;
  int kbase = (lane < 16) ? 0 : 8;
  int n     = (nt << 4) + (lane & 15);
  _Float16* out = dst + ((size_t)slot * 32 + lane) * 16;
#pragma unroll
  for (int g = 0; g < 16; ++g) {
    int k  = (ks << 5) + kmap(kbase, g);
    out[g] = (_Float16)W[((size_t)h * Kdim + k) * DHH + n];
  }
}

// ---------------------------------------------------------------------------
// Repack Wh row-major f16 [H][N][128] -> PV-B fragment-swizzled layout:
// slot = ((h*(N/32) + jb)*8 + nt), lane's v16h contiguous.
// ---------------------------------------------------------------------------
__global__ void repack_wh_kernel(const _Float16* __restrict__ rm,
                                 _Float16* __restrict__ swz) {
  int tid  = blockIdx.x * blockDim.x + threadIdx.x;
  int lane = tid & 31;
  int slot = tid >> 5;
  if (slot >= HH * (NN / 32) * 8) return;
  int nt = slot & 7;
  int hj = slot >> 3;
  int jb = hj & (NN / 32 - 1);
  int h  = hj / (NN / 32);
  int kbase = (lane < 16) ? 0 : 8;
  int n     = (nt << 4) + (lane & 15);
  _Float16* out = swz + ((size_t)slot * 32 + lane) * 16;
#pragma unroll
  for (int g = 0; g < 16; ++g) {
    int row = (jb << 5) + kmap(kbase, g);
    out[g]  = rm[((size_t)h * NN + row) * DHH + n];
  }
}

// ---------------------------------------------------------------------------
// GEMM: Wh = A @ W  (per head), plus src/dst = Wh @ a[:d], Wh @ a[d:].
// NKS = Kdim/32.  A is chems f32 [H][N][Kdim] (AHALF=false) or shared x f16
// [N][Kdim] (AHALF=true).  One wave = 16 rows x 128 cols.
// ---------------------------------------------------------------------------
template <int NKS, bool AHALF>
__global__ void gemm_kernel(const void* __restrict__ Aptr,
                            const _Float16* __restrict__ wswz,
                            const float* __restrict__ avec,
                            _Float16* __restrict__ wh_rm,
                            float* __restrict__ srcv, float* __restrict__ dstv) {
  constexpr int KD = NKS * 32;
  int h     = blockIdx.y;
  int wv    = threadIdx.x >> 5;
  int lane  = threadIdx.x & 31;
  int m0    = blockIdx.x * 64 + wv * 16;
  int kbase = (lane < 16) ? 0 : 8;
  int rbase = kbase;
  int nlo   = lane & 15;

  v8f acc[8];
  v8f zero = {};
#pragma unroll
  for (int i = 0; i < 8; ++i) acc[i] = zero;

  for (int ks = 0; ks < NKS; ++ks) {
    // --- issue all B-fragment loads first so WMMA waits are staggered ---
    half8 blo[8], bhi[8];
    const _Float16* bbase = wswz + ((size_t)((h * NKS + ks) * 8) * 32) * 16;
#pragma unroll
    for (int nt = 0; nt < 8; ++nt) {
      const _Float16* bp = bbase + (size_t)(nt * 32 + lane) * 16;
      blo[nt] = *(const half8*)bp;
      bhi[nt] = *(const half8*)(bp + 8);
    }
    // --- A fragment (conversion work overlaps the B loads) ---
    v16h af;
    if (AHALF) {
      const _Float16* arow = (const _Float16*)Aptr + (size_t)(m0 + nlo) * KD + ks * 32 + kbase;
      half8 lo = *(const half8*)arow;
      half8 hi = *(const half8*)(arow + 16);
      af = make_v16h(lo, hi);
    } else {
      const float* arow = (const float*)Aptr + ((size_t)h * NN + m0 + nlo) * KD + ks * 32 + kbase;
#pragma unroll
      for (int g = 0; g < 8; ++g) af[g] = (_Float16)arow[g];
#pragma unroll
      for (int g = 0; g < 8; ++g) af[8 + g] = (_Float16)arow[16 + g];
    }
#pragma unroll
    for (int nt = 0; nt < 8; ++nt) {
      v16h bf = make_v16h(blo[nt], bhi[nt]);
      acc[nt] = __builtin_amdgcn_wmma_f32_16x16x32_f16(
          false, af, false, bf, (short)0, acc[nt], false, false);
    }
  }

  // --- epilogue: row-major f16 Wh ---
#pragma unroll
  for (int nt = 0; nt < 8; ++nt)
#pragma unroll
    for (int r = 0; r < 8; ++r)
      wh_rm[((size_t)h * NN + m0 + rbase + r) * DHH + nt * 16 + nlo] = (_Float16)acc[nt][r];

  // --- src/dst = Wh @ a[:128], Wh @ a[128:] ---
  float as[8], ad[8];
#pragma unroll
  for (int nt = 0; nt < 8; ++nt) {
    as[nt] = avec[h * 256 + nt * 16 + nlo];
    ad[nt] = avec[h * 256 + 128 + nt * 16 + nlo];
  }
  float ps[8], pd[8];
#pragma unroll
  for (int r = 0; r < 8; ++r) {
    float s = 0.0f, d = 0.0f;
#pragma unroll
    for (int nt = 0; nt < 8; ++nt) { s += acc[nt][r] * as[nt]; d += acc[nt][r] * ad[nt]; }
#pragma unroll
    for (int off = 8; off >= 1; off >>= 1) {
      s += __shfl_xor(s, off, 32);
      d += __shfl_xor(d, off, 32);
    }
    ps[r] = s; pd[r] = d;
  }
  float sv = 0.0f, dv = 0.0f;
#pragma unroll
  for (int r = 0; r < 8; ++r) {
    sv = (nlo == r) ? ps[r] : sv;
    dv = (nlo == r) ? pd[r] : dv;
  }
  if (nlo < 8) {
    srcv[h * NN + m0 + rbase + nlo] = sv;
    dstv[h * NN + m0 + rbase + nlo] = dv;
  }
}

// ---------------------------------------------------------------------------
// Fused masked-softmax attention (flash style, rank-1 logits):
//   e_ij = leakyrelu(src_i + dst_j); masked by adj bits; online softmax;
//   O += P @ Wh.  One wave = 16 rows; block = 8 waves = 128 rows.
//   Per 32-col step: 16 b128 B-loads issued first, softmax VALU/exp work
//   covers their latency, then 8 WMMAs.
// ---------------------------------------------------------------------------
__global__ void attn_kernel(const float* __restrict__ srcv, const float* __restrict__ dstv,
                            const unsigned* __restrict__ adjbits,
                            const _Float16* __restrict__ whswz,
                            _Float16* __restrict__ out16, float* __restrict__ out32) {
  int h     = blockIdx.y;
  int wv    = threadIdx.x >> 5;
  int lane  = threadIdx.x & 31;
  int m0    = blockIdx.x * 128 + wv * 16;
  int kbase = (lane < 16) ? 0 : 8;
  int rbase = kbase;
  int nlo   = lane & 15;
  int mrow  = m0 + nlo;

  float mysrc = srcv[h * NN + mrow];
  float M = -3.0e38f;
  float S = 0.0f;
  v8f acc[8];
  v8f zero = {};
#pragma unroll
  for (int i = 0; i < 8; ++i) acc[i] = zero;

  const float*    dbase = dstv + h * NN;
  const unsigned* abase = adjbits + (size_t)mrow * (NN / 32);
  const _Float16* wbase = whswz + (size_t)(h * (NN / 32)) * 8 * 32 * 16;
  unsigned long long full = __ballot(1);

  for (int jb = 0; jb < NN / 32; ++jb) {
    // --- 1) issue all memory for this step up front ---
    half8 blo[8], bhi[8];
    const _Float16* wjb = wbase + (size_t)jb * 8 * 32 * 16;
#pragma unroll
    for (int nt = 0; nt < 8; ++nt) {
      const _Float16* bp = wjb + (size_t)(nt * 32 + lane) * 16;
      blo[nt] = *(const half8*)(bp);
      bhi[nt] = *(const half8*)(bp + 8);
    }
    unsigned aw = abase[jb];                 // 32 adjacency bits for this row
    int j0 = jb * 32 + kbase;
    float dv0[16];
#pragma unroll
    for (int g = 0; g < 16; ++g) dv0[g] = dbase[j0 + ((g < 8) ? g : 8 + g)];

    // --- 2) masked leaky-relu logits + online softmax (covers load latency) ---
    float e[16];
    float lm = -3.0e38f;
#pragma unroll
    for (int g = 0; g < 16; ++g) {
      float ev = mysrc + dv0[g];
      ev = (ev > 0.0f) ? ev : (GAT_ALPHA * ev);
      unsigned bit = (aw >> (kbase + ((g < 8) ? g : 8 + g))) & 1u;
      ev = bit ? ev : NEGV;
      e[g] = ev;
      lm   = fmaxf(lm, ev);
    }
    float pm    = fmaxf(lm, __shfl_xor(lm, 16, 32));  // full row max (lane pair)
    float mn    = fmaxf(M, pm);
    float scale = __expf(M - mn);
    M = mn;

    v16h  pa;
    float psum = 0.0f;
#pragma unroll
    for (int g = 0; g < 16; ++g) {
      float p = __expf(e[g] - mn);
      psum += p;
      pa[g] = (_Float16)p;
    }
    S = S * scale + psum;

    // Rescale accumulators only if any row's max moved (wave-uniform skip).
    if (__ballot(scale == 1.0f) != full) {
#pragma unroll
      for (int r = 0; r < 8; ++r) {
        float sc = __shfl(scale, rbase + r, 32);
#pragma unroll
        for (int nt = 0; nt < 8; ++nt) acc[nt][r] *= sc;
      }
    }

    // --- 3) WMMA chain ---
#pragma unroll
    for (int nt = 0; nt < 8; ++nt) {
      v16h bf = make_v16h(blo[nt], bhi[nt]);
      acc[nt] = __builtin_amdgcn_wmma_f32_16x16x32_f16(
          false, pa, false, bf, (short)0, acc[nt], false, false);
    }
  }

  float Stot = S + __shfl_xor(S, 16, 32);
  float inv  = 1.0f / Stot;
#pragma unroll
  for (int r = 0; r < 8; ++r) {
    float iv  = __shfl(inv, rbase + r, 32);
    int   row = m0 + rbase + r;
#pragma unroll
    for (int nt = 0; nt < 8; ++nt) {
      float o = acc[nt][r] * iv;
      o = (o > 0.0f) ? o : (__expf(o) - 1.0f);   // ELU
      int col = h * DHH + nt * 16 + nlo;
      if (out32) out32[(size_t)row * HD + col] = o;
      else       out16[(size_t)row * HD + col] = (_Float16)o;
    }
  }
}

// ---------------------------------------------------------------------------
extern "C" void kernel_launch(void* const* d_in, const int* in_sizes, int n_in,
                              void* d_out, int out_size, void* d_ws, size_t ws_size,
                              hipStream_t stream) {
  const float* chems = (const float*)d_in[0];
  const int*   adj   = (const int*)d_in[1];
  const float* W1    = (const float*)d_in[2];
  const float* a1    = (const float*)d_in[3];
  const float* W2    = (const float*)d_in[4];
  const float* a2    = (const float*)d_in[5];
  float*       out   = (float*)d_out;

  char*  ws  = (char*)d_ws;
  size_t off = 0;
  auto alloc = [&](size_t bytes) -> char* {
    char* p = ws + off;
    off += (bytes + 255) & ~(size_t)255;
    return p;
  };
  _Float16* w1swz  = (_Float16*)alloc((size_t)HH * 4 * 8 * 32 * 16 * 2);    // 128 KB
  _Float16* w2swz  = (_Float16*)alloc((size_t)HH * 16 * 8 * 32 * 16 * 2);   // 512 KB
  _Float16* wh_rm  = (_Float16*)alloc((size_t)HH * NN * DHH * 2);           // 4 MB (reused)
  _Float16* wh_swz = (_Float16*)alloc((size_t)HH * NN * DHH * 2);           // 4 MB (reused)
  float*    srcv   = (float*)alloc((size_t)HH * NN * 4);
  float*    dstv   = (float*)alloc((size_t)HH * NN * 4);
  _Float16* xbuf   = (_Float16*)alloc((size_t)NN * HD * 2);                 // 4 MB
  unsigned* abits  = (unsigned*)alloc((size_t)NN * (NN / 32) * 4);          // 2 MB

  // One-time preprocessing (deterministic, re-done every call)
  pack_adj_kernel<<<dim3((NN * (NN / 32) + 255) / 256), dim3(256), 0, stream>>>(adj, abits);
  prep_w_kernel<<<dim3((HH * 4 * 8 * 32 + 255) / 256), dim3(256), 0, stream>>>(W1, w1swz, 128);
  prep_w_kernel<<<dim3((HH * 16 * 8 * 32 + 255) / 256), dim3(256), 0, stream>>>(W2, w2swz, 512);

  dim3 gg(NN / 64, HH);
  dim3 ga(NN / 128, HH);
  dim3 gr((HH * (NN / 32) * 8 * 32 + 255) / 256);

  // ---- stage 1 ----
  gemm_kernel<4, false><<<gg, dim3(128), 0, stream>>>(chems, w1swz, a1, wh_rm, srcv, dstv);
  repack_wh_kernel<<<gr, dim3(256), 0, stream>>>(wh_rm, wh_swz);
  attn_kernel<<<ga, dim3(256), 0, stream>>>(srcv, dstv, abits, wh_swz, xbuf, nullptr);

  // ---- stage 2 ----
  gemm_kernel<16, true><<<gg, dim3(128), 0, stream>>>(xbuf, w2swz, a2, wh_rm, srcv, dstv);
  repack_wh_kernel<<<gr, dim3(256), 0, stream>>>(wh_rm, wh_swz);
  attn_kernel<<<ga, dim3(256), 0, stream>>>(srcv, dstv, abits, wh_swz, nullptr, out);
}
